// Model_24824910971198
// MI455X (gfx1250) — matrix operand order repaired and checked by reading, not compiled, not run on hardware
//
#include <hip/hip_runtime.h>
#include <hip/hip_bf16.h>
#include <math.h>

typedef __attribute__((ext_vector_type(16))) _Float16 v16h;
typedef __attribute__((ext_vector_type(8)))  _Float16 v8h;
typedef __attribute__((ext_vector_type(8)))  float    v8f;

#define B_    2
#define SEQ   4096
#define C_    16
#define S_    32
#define NSEG  128          // N = SEQ/S
#define L2_   256          // 2N tokens per sequence
#define D_    256
#define NHEAD 8
#define DH    32
#define BC_   (B_*C_)      // 32 sequences
#define NTOK  (BC_*L2_)    // 8192 tokens

#define MODE_F16       0
#define MODE_F16_GELU  1
#define MODE_RESID     2

// ---------------------------------------------------------------- utilities
__device__ __forceinline__ float block_reduce_sum(float v, float* red) {
    int t = threadIdx.x;
    red[t] = v;
    __syncthreads();
    for (int s = 128; s > 0; s >>= 1) {
        if (t < s) red[t] += red[t + s];
        __syncthreads();
    }
    float r = red[0];
    __syncthreads();
    return r;
}

__global__ __launch_bounds__(256) void cvt_f16_kernel(const float* __restrict__ src,
                                                      _Float16* __restrict__ dst, int n) {
    int i = blockIdx.x * 256 + threadIdx.x;
    if (i < n) dst[i] = (_Float16)src[i];
}

// ------------------------------------------------------------- input stats
// mean over last 32 samples, std (ddof=1) over all 4096, per (b,c)
__global__ __launch_bounds__(256) void stats_kernel(const float* __restrict__ x,
                                                    float* __restrict__ meanp,
                                                    float* __restrict__ stdp) {
    __shared__ float red[256];
    int bc = blockIdx.x, b = bc >> 4, c = bc & 15;
    int t = threadIdx.x;
    float s = 0.f, sq = 0.f;
    for (int l = t; l < SEQ; l += 256) {
        float v = x[((size_t)b * SEQ + l) * C_ + c];
        s += v; sq += v * v;
    }
    float tot   = block_reduce_sum(s, red);
    float totsq = block_reduce_sum(sq, red);
    float ls = (t < S_) ? x[((size_t)b * SEQ + (SEQ - S_) + t) * C_ + c] : 0.f;
    float lastsum = block_reduce_sum(ls, red);
    if (t == 0) {
        meanp[bc] = lastsum / (float)S_;
        float var = (totsq - tot * tot / (float)SEQ) / (float)(SEQ - 1);
        stdp[bc]  = sqrtf(var) + 1e-8f;
    }
}

// ------------------------------------------------- embedding (fused, VALU)
// one block per (segment n, bc); produces h[bc][2n] (ctx) and h[bc][2n+1] (main)
__global__ __launch_bounds__(256) void embed_kernel(
    const float* __restrict__ x, const float* __restrict__ meanp, const float* __restrict__ stdp,
    const float* __restrict__ tok_w, const float* __restrict__ tok_b,
    const float* __restrict__ ccp_w, const float* __restrict__ ccp_b,
    const float* __restrict__ ctp_w, const float* __restrict__ ctp_b,
    const float* __restrict__ ncw, const float* __restrict__ nmw,
    const float* __restrict__ pnw, const float* __restrict__ pos,
    const float* __restrict__ cemb, float* __restrict__ h) {
    __shared__ float seg[C_][S_];
    __shared__ float mixed[S_];
    __shared__ float red[256];
    int n = blockIdx.x, bc = blockIdx.y;
    int b = bc >> 4, t = bc & 15;
    int d = threadIdx.x;

    for (int i = d; i < C_ * S_; i += 256) {
        int c = i >> 5, s = i & 31;
        float mu = meanp[b * C_ + c], sd = stdp[b * C_ + c];
        seg[c][s] = (x[((size_t)b * SEQ + n * S_ + s) * C_ + c] - mu) / sd;
    }
    __syncthreads();
    if (d < S_) {
        float acc = ccp_b[t];
        #pragma unroll
        for (int c = 0; c < C_; ++c) acc += seg[c][d] * ccp_w[t * C_ + c];
        mixed[d] = acc;
    }
    __syncthreads();

    float ctx = ctp_b[d], mn = tok_b[d];
    #pragma unroll
    for (int s = 0; s < S_; ++s) {
        ctx += mixed[s] * ctp_w[d * S_ + s];
        mn  += seg[t][s] * tok_w[d * S_ + s];
    }
    float ssc = block_reduce_sum(ctx * ctx, red);
    float ssm = block_reduce_sum(mn * mn, red);
    ctx = ctx * rsqrtf(ssc / (float)D_ + 1e-5f) * ncw[d];
    mn  = mn  * rsqrtf(ssm / (float)D_ + 1e-5f) * nmw[d];

    int l0 = 2 * n, l1 = 2 * n + 1;
    ctx += pos[l0 * D_ + d] + cemb[((size_t)(t * L2_) + l0) * D_ + d];
    mn  += pos[l1 * D_ + d] + cemb[((size_t)(t * L2_) + l1) * D_ + d];

    float s2c = block_reduce_sum(ctx * ctx, red);
    float s2m = block_reduce_sum(mn * mn, red);
    ctx = ctx * rsqrtf(s2c / (float)D_ + 1e-5f) * pnw[d];
    mn  = mn  * rsqrtf(s2m / (float)D_ + 1e-5f) * pnw[d];

    size_t base = (size_t)bc * L2_ * D_;
    h[base + (size_t)l0 * D_ + d] = ctx;
    h[base + (size_t)l1 * D_ + d] = mn;
}

// ---------------------------------------------------------------- rmsnorm
__global__ __launch_bounds__(256) void rmsnorm_kernel(const float* __restrict__ h,
                                                      const float* __restrict__ wgt,
                                                      _Float16* __restrict__ z) {
    __shared__ float red[256];
    size_t row = blockIdx.x;
    int d = threadIdx.x;
    float v = h[row * D_ + d];
    float ss = block_reduce_sum(v * v, red);
    z[row * D_ + d] = (_Float16)(v * rsqrtf(ss / (float)D_ + 1e-5f) * wgt[d]);
}

// ------------------------------------------------------------- WMMA GEMM
// C[m][n] = sum_k A[m][k] * W[n][k] (+bias, epilogue by mode)
// register-blocked: each wave computes a 64x16 strip (4 M-tiles, 1 N-tile),
// reusing the B (weight) fragment across 4 WMMAs per k-step.
__device__ __forceinline__ v16h load_afrag(const _Float16* p) {
    v8h a0 = *(const v8h*)(p);
    v8h a1 = *(const v8h*)(p + 16);
    return __builtin_shufflevector(a0, a1, 0, 1, 2, 3, 4, 5, 6, 7,
                                           8, 9, 10, 11, 12, 13, 14, 15);
}

__global__ __launch_bounds__(128) void gemm_wmma(
    const _Float16* __restrict__ A, const _Float16* __restrict__ W,
    const float* __restrict__ bias, void* __restrict__ outp,
    int N, int K, int mode) {
    int wave = threadIdx.x >> 5;
    int lane = threadIdx.x & 31;
    int mbase = blockIdx.x * 64;
    int nbase = (blockIdx.y * 4 + wave) * 16;
    int lh = lane & 15;
    int hi = (lane >> 4) & 1;

    // A 16x32 f16 lane layout: lanes 0-15 hold K 0..7 & 16..23; lanes 16-31 hold 8..15 & 24..31
    const _Float16* Ar = A + (size_t)(mbase + lh) * K + (hi ? 8 : 0);
    // B 32x16 f16 lane layout: lanes 0-15 hold K 0..7 & 16..23; lanes 16-31 hold 8..15 & 24..31
    const _Float16* Wr = W + (size_t)(nbase + lh) * K + (hi ? 8 : 0);
    const size_t mstride = (size_t)16 * K;

    v8f c0 = {}, c1 = {}, c2 = {}, c3 = {};
    for (int k0 = 0; k0 < K; k0 += 32) {
        v16h b  = load_afrag(Wr + k0);
        v16h a0 = load_afrag(Ar + k0);
        v16h a1 = load_afrag(Ar + mstride + k0);
        v16h a2 = load_afrag(Ar + 2 * mstride + k0);
        v16h a3 = load_afrag(Ar + 3 * mstride + k0);
        c0 = __builtin_amdgcn_wmma_f32_16x16x32_f16(false, a0, false, b, (short)0, c0, false, false);
        c1 = __builtin_amdgcn_wmma_f32_16x16x32_f16(false, a1, false, b, (short)0, c1, false, false);
        c2 = __builtin_amdgcn_wmma_f32_16x16x32_f16(false, a2, false, b, (short)0, c2, false, false);
        c3 = __builtin_amdgcn_wmma_f32_16x16x32_f16(false, a3, false, b, (short)0, c3, false, false);
    }

    int n = nbase + lh;
    float bn = bias[n];
    v8f acc[4] = {c0, c1, c2, c3};
    #pragma unroll
    for (int tIdx = 0; tIdx < 4; ++tIdx) {
        int m0 = mbase + tIdx * 16 + hi * 8;
        if (mode == MODE_RESID) {
            float* out = (float*)outp;
            #pragma unroll
            for (int r = 0; r < 8; ++r) out[(size_t)(m0 + r) * N + n] += acc[tIdx][r] + bn;
        } else {
            _Float16* out = (_Float16*)outp;
            #pragma unroll
            for (int r = 0; r < 8; ++r) {
                float v = acc[tIdx][r] + bn;
                if (mode == MODE_F16_GELU)
                    v = 0.5f * v * (1.f + erff(v * 0.70710678118654752f));
                out[(size_t)(m0 + r) * N + n] = (_Float16)v;
            }
        }
    }
}

// ----------------------------------------------- causal linear attention
// one wave per (sequence, head); per-lane column e of 32x32 KV state in regs
__global__ __launch_bounds__(256) void attn_kernel(const _Float16* __restrict__ q,
                                                   const _Float16* __restrict__ k,
                                                   const _Float16* __restrict__ v,
                                                   _Float16* __restrict__ a) {
    int bc   = blockIdx.x;
    int head = threadIdx.x >> 5;
    int lane = threadIdx.x & 31;
    size_t base = ((size_t)bc * L2_) * D_ + head * DH + lane;

    float kv[DH];
    #pragma unroll
    for (int d = 0; d < DH; ++d) kv[d] = 0.f;

    for (int l = 0; l < L2_; ++l) {
        size_t idx = base + (size_t)l * D_;
        float kl = (float)k[idx];
        float vl = (float)v[idx];
        float ql = (float)q[idx];
        float acc = 0.f;
        #pragma unroll
        for (int d = 0; d < DH; ++d) {
            float kd = __shfl(kl, d, 32);
            float qd = __shfl(ql, d, 32);
            kv[d] += kd * vl;          // inclusive cumsum of k x v^T
            acc   += qd * kv[d];
        }
        a[idx] = (_Float16)acc;
    }
}

// ------------------------------------- final norm + output head + denorm
__global__ __launch_bounds__(256) void final_kernel(
    const float* __restrict__ h, const float* __restrict__ fw,
    const float* __restrict__ out_w, const float* __restrict__ out_b,
    const float* __restrict__ meanp, const float* __restrict__ stdp,
    float* __restrict__ out) {
    __shared__ float red[256];
    __shared__ float nz[256];
    int bc = blockIdx.x;
    int d = threadIdx.x;
    const float* hr = h + ((size_t)bc * L2_ + (L2_ - 1)) * D_;   // last main token
    float v = hr[d];
    float ss = block_reduce_sum(v * v, red);
    nz[d] = v * rsqrtf(ss / (float)D_ + 1e-5f) * fw[d];
    __syncthreads();
    if (d < S_) {
        float acc = out_b[d];
        for (int j = 0; j < D_; ++j) acc += nz[j] * out_w[d * D_ + j];
        float val = acc * stdp[bc] + meanp[bc];
        int b = bc >> 4, t = bc & 15;
        out[((size_t)b * S_ + d) * C_ + t] = val;   // (B, PRED_LEN, C)
    }
}

// ---------------------------------------------------------------- driver
extern "C" void kernel_launch(void* const* d_in, const int* in_sizes, int n_in,
                              void* d_out, int out_size, void* d_ws, size_t ws_size,
                              hipStream_t stream) {
    (void)in_sizes; (void)n_in; (void)out_size; (void)ws_size;
    const float* x       = (const float*)d_in[0];
    const float* tok_w   = (const float*)d_in[1];
    const float* tok_b   = (const float*)d_in[2];
    const float* ccp_w   = (const float*)d_in[3];
    const float* ccp_b   = (const float*)d_in[4];
    const float* ctp_w   = (const float*)d_in[5];
    const float* ctp_b   = (const float*)d_in[6];
    const float* out_w   = (const float*)d_in[7];
    const float* out_b   = (const float*)d_in[8];
    const float* norm_main_w = (const float*)d_in[9];
    const float* norm_ctx_w  = (const float*)d_in[10];
    const float* pre_norm_w  = (const float*)d_in[11];
    const float* final_norm_w= (const float*)d_in[12];
    const float* pos_emb = (const float*)d_in[13];
    const float* ch_emb  = (const float*)d_in[14];
    const float* q_w     = (const float*)d_in[15];
    const float* q_b     = (const float*)d_in[16];
    const float* k_w     = (const float*)d_in[17];
    const float* k_b     = (const float*)d_in[18];
    const float* v_w     = (const float*)d_in[19];
    const float* v_b     = (const float*)d_in[20];
    const float* cproj_w = (const float*)d_in[21];
    const float* cproj_b = (const float*)d_in[22];
    const float* pn_w    = (const float*)d_in[23];
    const float* ln_w    = (const float*)d_in[24];
    const float* fc_w    = (const float*)d_in[25];
    const float* fc_b    = (const float*)d_in[26];
    const float* proj_w  = (const float*)d_in[27];
    const float* proj_b  = (const float*)d_in[28];

    char* wsb = (char*)d_ws;
    size_t off = 0;
    auto take = [&](size_t bytes) -> void* {
        void* p = wsb + off;
        off = (off + bytes + 255) & ~(size_t)255;
        return p;
    };
    float*    meanp = (float*)take(BC_ * 4);
    float*    stdp  = (float*)take(BC_ * 4);
    float*    h     = (float*)take((size_t)NTOK * D_ * 4);
    _Float16* z16   = (_Float16*)take((size_t)NTOK * D_ * 2);
    _Float16* q16   = (_Float16*)take((size_t)NTOK * D_ * 2);
    _Float16* k16   = (_Float16*)take((size_t)NTOK * D_ * 2);
    _Float16* v16   = (_Float16*)take((size_t)NTOK * D_ * 2);
    _Float16* a16   = (_Float16*)take((size_t)NTOK * D_ * 2);
    _Float16* fc16  = (_Float16*)take((size_t)NTOK * 4 * D_ * 2);
    _Float16* wq16  = (_Float16*)take((size_t)2 * D_ * D_ * 2);
    _Float16* wk16  = (_Float16*)take((size_t)2 * D_ * D_ * 2);
    _Float16* wv16  = (_Float16*)take((size_t)2 * D_ * D_ * 2);
    _Float16* wc16  = (_Float16*)take((size_t)2 * D_ * D_ * 2);
    _Float16* wfc16 = (_Float16*)take((size_t)2 * 4 * D_ * D_ * 2);
    _Float16* wp16  = (_Float16*)take((size_t)2 * D_ * 4 * D_ * 2);

    const int nQKV = 2 * D_ * D_;       // 131072
    const int nFC  = 2 * 4 * D_ * D_;   // 524288
    cvt_f16_kernel<<<(nQKV + 255) / 256, 256, 0, stream>>>(q_w, wq16, nQKV);
    cvt_f16_kernel<<<(nQKV + 255) / 256, 256, 0, stream>>>(k_w, wk16, nQKV);
    cvt_f16_kernel<<<(nQKV + 255) / 256, 256, 0, stream>>>(v_w, wv16, nQKV);
    cvt_f16_kernel<<<(nQKV + 255) / 256, 256, 0, stream>>>(cproj_w, wc16, nQKV);
    cvt_f16_kernel<<<(nFC + 255) / 256, 256, 0, stream>>>(fc_w, wfc16, nFC);
    cvt_f16_kernel<<<(nFC + 255) / 256, 256, 0, stream>>>(proj_w, wp16, nFC);

    stats_kernel<<<BC_, 256, 0, stream>>>(x, meanp, stdp);
    embed_kernel<<<dim3(NSEG, BC_), 256, 0, stream>>>(
        x, meanp, stdp, tok_w, tok_b, ccp_w, ccp_b, ctp_w, ctp_b,
        norm_ctx_w, norm_main_w, pre_norm_w, pos_emb, ch_emb, h);

    dim3 gD(NTOK / 64, D_ / 64);        // 128 x 4
    dim3 gFC(NTOK / 64, (4 * D_) / 64); // 128 x 16
    for (int i = 0; i < 2; ++i) {
        rmsnorm_kernel<<<NTOK, 256, 0, stream>>>(h, pn_w + i * D_, z16);
        gemm_wmma<<<gD, 128, 0, stream>>>(z16, wq16 + (size_t)i * D_ * D_, q_b + i * D_, q16, D_, D_, MODE_F16);
        gemm_wmma<<<gD, 128, 0, stream>>>(z16, wk16 + (size_t)i * D_ * D_, k_b + i * D_, k16, D_, D_, MODE_F16);
        gemm_wmma<<<gD, 128, 0, stream>>>(z16, wv16 + (size_t)i * D_ * D_, v_b + i * D_, v16, D_, D_, MODE_F16);
        attn_kernel<<<BC_, 256, 0, stream>>>(q16, k16, v16, a16);
        gemm_wmma<<<gD, 128, 0, stream>>>(a16, wc16 + (size_t)i * D_ * D_, cproj_b + i * D_, h, D_, D_, MODE_RESID);
        rmsnorm_kernel<<<NTOK, 256, 0, stream>>>(h, ln_w + i * D_, z16);
        gemm_wmma<<<gFC, 128, 0, stream>>>(z16, wfc16 + (size_t)i * 4 * D_ * D_, fc_b + i * 4 * D_, fc16, 4 * D_, D_, MODE_F16_GELU);
        gemm_wmma<<<gD, 128, 0, stream>>>(fc16, wp16 + (size_t)i * D_ * 4 * D_, proj_b + i * D_, h, D_, 4 * D_, MODE_RESID);
    }
    final_kernel<<<BC_, 256, 0, stream>>>(h, final_norm_w, out_w, out_b, meanp, stdp, (float*)d_out);
}
